// SparseMoE_52261162057764
// MI455X (gfx1250) — compile-verified, matrix operand order checked by reading, not run on hardware
//
#include <hip/hip_runtime.h>
#include <hip/hip_bf16.h>
#include <math.h>

// Sizes fixed by the reference problem.
#define T_TOK 8192
#define H_DIM 1024
#define I_DIM 4096
#define N_EXP 8
#define TM    32     // tokens per FFN tile (2 row-groups of 16)
#define KC    256    // I-chunk width (16 waves x 16 columns)
#define NTHR  512    // 16 waves per block
#define WAVES 8      // router: waves per block

typedef __attribute__((ext_vector_type(16))) __bf16 v16bf;
typedef __attribute__((ext_vector_type(8)))  __bf16 v8bf;
typedef __attribute__((ext_vector_type(8)))  float  v8f;

// B operand from fp32 weights (fallback path): 8 VALU pk-converts.
__device__ __forceinline__ v16bf mk_b(const float* __restrict__ p) {
    v16bf b;
#pragma unroll
    for (int i = 0; i < 8; ++i) { b[i] = (__bf16)p[i]; b[i + 8] = (__bf16)p[16 + i]; }
    return b;
}
// B operand from preconverted bf16 weights: two 16B global loads, no VALU.
__device__ __forceinline__ v16bf mk_b(const __bf16* __restrict__ p) {
    v8bf lo = *(const v8bf*)(p);
    v8bf hi = *(const v8bf*)(p + 16);
    return __builtin_shufflevector(lo, hi, 0, 1, 2, 3, 4, 5, 6, 7,
                                   8, 9, 10, 11, 12, 13, 14, 15);
}
// A operand from bf16 LDS: two ds_load_b128.
__device__ __forceinline__ v16bf lds_a16(const __bf16* __restrict__ p) {
    v8bf lo = *(const v8bf*)(p);
    v8bf hi = *(const v8bf*)(p + 16);
    return __builtin_shufflevector(lo, hi, 0, 1, 2, 3, 4, 5, 6, 7,
                                   8, 9, 10, 11, 12, 13, 14, 15);
}

// ---------------------------------------------------------------------------
// Kernel 0: zero MoE output region + expert counters (atomic accum needs 0s).
// ---------------------------------------------------------------------------
__global__ void moe_zero_kernel(float* __restrict__ out, int* __restrict__ counts) {
    int i = blockIdx.x * blockDim.x + threadIdx.x;
    if (i < N_EXP) counts[i] = 0;
    float4 z = {0.f, 0.f, 0.f, 0.f};
    *(float4*)(out + (size_t)i * 4) = z;   // grid sized so i*4 < T*H exactly
}

// ---------------------------------------------------------------------------
// Kernel 0b: fp32 -> bf16 weight conversion (one pass; result lives in L2).
// ---------------------------------------------------------------------------
__global__ void moe_cvt_kernel(const float* __restrict__ src, __bf16* __restrict__ dst) {
    size_t base = ((size_t)blockIdx.x * blockDim.x + threadIdx.x) * 8;
    float4 a = *(const float4*)(src + base);
    float4 b = *(const float4*)(src + base + 4);
    v8bf o;
    o[0] = (__bf16)a.x; o[1] = (__bf16)a.y; o[2] = (__bf16)a.z; o[3] = (__bf16)a.w;
    o[4] = (__bf16)b.x; o[5] = (__bf16)b.y; o[6] = (__bf16)b.z; o[7] = (__bf16)b.w;
    *(v8bf*)(dst + base) = o;
}

// ---------------------------------------------------------------------------
// Kernel 1: router. One wave (32 lanes) per token.
// ---------------------------------------------------------------------------
__global__ void moe_router_kernel(const float* __restrict__ x,
                                  const float* __restrict__ wg,
                                  float* __restrict__ logits_out,
                                  int*   __restrict__ counts,
                                  int*   __restrict__ tok_list,
                                  float* __restrict__ gate_list) {
    const int lane = threadIdx.x & 31;
    const int wave = threadIdx.x >> 5;
    const int t = blockIdx.x * WAVES + wave;
    const float* xr = x + (size_t)t * H_DIM;

    float acc[N_EXP];
#pragma unroll
    for (int e = 0; e < N_EXP; ++e) acc[e] = 0.f;

    for (int h = lane; h < H_DIM; h += 32) {
        float xv = xr[h];
#pragma unroll
        for (int e = 0; e < N_EXP; ++e) acc[e] += xv * wg[e * H_DIM + h];
    }
#pragma unroll
    for (int e = 0; e < N_EXP; ++e) {
#pragma unroll
        for (int m = 16; m >= 1; m >>= 1) acc[e] += __shfl_xor(acc[e], m, 32);
    }

    if (lane == 0) {
        float b0 = -INFINITY, b1 = -INFINITY;
        int   i0 = 0, i1 = 0;
#pragma unroll
        for (int e = 0; e < N_EXP; ++e) {
            logits_out[(size_t)t * N_EXP + e] = acc[e];
            float v = acc[e];
            if (v > b0)      { b1 = b0; i1 = i0; b0 = v; i0 = e; }
            else if (v > b1) { b1 = v;  i1 = e; }
        }
        float ex = expf(b1 - b0);            // b1 <= b0, safe
        float g0 = 1.f / (1.f + ex);
        float g1 = ex  / (1.f + ex);
        int s0 = atomicAdd(&counts[i0], 1);
        tok_list [i0 * T_TOK + s0] = t;
        gate_list[i0 * T_TOK + s0] = g0;
        int s1 = atomicAdd(&counts[i1], 1);
        tok_list [i1 * T_TOK + s1] = t;
        gate_list[i1 * T_TOK + s1] = g1;
    }
}

// ---------------------------------------------------------------------------
// Kernel 2: fused expert FFN (fc -> gelu -> proj), bf16 WMMA, f32 accum.
// Block = 16 waves, one 32-token tile of one expert.
//   phase 1: as[32 x 256] = gelu(x_tile[32x1024] @ w_fc_chunk[256x1024]^T)
//            wave w owns I-columns [16w,16w+16): one B fetch -> 2 WMMAs
//            (row-groups 0-15 / 16-31), no duplicated weight loads.
//   phase 2: acc[32 x 1024] += as @ w_proj[:, chunk]^T
//            wave w owns H-columns [64w,64w+64): 4 col-tiles x 2 row-groups,
//            one B fetch -> 2 WMMAs; 8 persistent v8f accumulators.
// ---------------------------------------------------------------------------
template <typename WT>
__global__ void __launch_bounds__(NTHR)
moe_ffn_kernel(const float* __restrict__ x,
               const WT*    __restrict__ w_fc,
               const WT*    __restrict__ w_proj,
               const int*   __restrict__ counts,
               const int*   __restrict__ tok_list,
               const float* __restrict__ gate_list,
               float* __restrict__ out) {
    __shared__ __bf16 xs[TM][H_DIM];   // 64 KB: gathered x tile (bf16)
    __shared__ __bf16 as[TM][KC];      // 16 KB: gelu(fc) chunk (bf16)
    __shared__ int    tok_s[TM];
    __shared__ float  gate_s[TM];

    const int e    = blockIdx.y;
    const int tile = blockIdx.x;
    const int cnt  = counts[e];
    if (tile * TM >= cnt) return;      // block-uniform: EXEC stays all-ones

    const int tid  = threadIdx.x;
    const int lane = tid & 31;
    const int wave = tid >> 5;         // 0..15
    const int hs8  = (lane >> 4) * 8;  // half-select * 8
    const int ln   = lane & 15;

    if (tid < TM) {
        int idx = tile * TM + tid;
        if (idx < cnt) { tok_s[tid] = tok_list[e * T_TOK + idx];
                         gate_s[tid] = gate_list[e * T_TOK + idx]; }
        else           { tok_s[tid] = 0; gate_s[tid] = 0.f; }
    }
    __syncthreads();

    // Gather token rows, convert fp32 -> bf16 into LDS (coalesced along H).
    for (int i = tid; i < TM * H_DIM; i += NTHR) {
        int r = i >> 10, c = i & (H_DIM - 1);
        xs[r][c] = (__bf16)x[(size_t)tok_s[r] * H_DIM + c];
    }
    __syncthreads();

    const WT* wfc_e = w_fc   + (size_t)e * I_DIM * H_DIM;
    const WT* wpr_e = w_proj + (size_t)e * H_DIM * I_DIM;

    v8f acc[4][2];                     // [col-tile][row-group]
#pragma unroll
    for (int t = 0; t < 4; ++t) { acc[t][0] = (v8f){}; acc[t][1] = (v8f){}; }

    for (int cb = 0; cb < I_DIM; cb += KC) {
        // ---- phase 1: wave owns w_fc rows [cb+16*wave, cb+16*wave+16)
        const WT* brow = wfc_e + (size_t)(cb + wave * 16 + ln) * H_DIM;
        if (cb + KC < I_DIM)  // hint next chunk's w_fc rows -> global_prefetch
            __builtin_prefetch(wfc_e + (size_t)(cb + KC + wave * 16 + ln) * H_DIM, 0, 1);

        v8f c0 = (v8f){}, c1 = (v8f){};
#pragma unroll
        for (int k = 0; k < H_DIM / 32; ++k) {
            v16bf b  = mk_b(brow + k * 32 + hs8);
            v16bf a0 = lds_a16(&xs[ln][k * 32 + hs8]);        // rows 0..15
            v16bf a1 = lds_a16(&xs[16 + ln][k * 32 + hs8]);   // rows 16..31
            c0 = __builtin_amdgcn_wmma_f32_16x16x32_bf16(false, a0, false, b,
                                                         (short)0, c0, false, false);
            c1 = __builtin_amdgcn_wmma_f32_16x16x32_bf16(false, a1, false, b,
                                                         (short)0, c1, false, false);
        }
        __syncthreads();   // prior phase-2 reads of `as` are complete
#pragma unroll
        for (int r = 0; r < 8; ++r) {
            float v0 = c0[r], v1 = c1[r];
            float g0 = 0.5f * v0 * (1.0f + erff(v0 * 0.70710678118654752f));
            float g1 = 0.5f * v1 * (1.0f + erff(v1 * 0.70710678118654752f));
            as[r + hs8][wave * 16 + ln]      = (__bf16)g0;  // D: M=r+hs*8, N=ln
            as[16 + r + hs8][wave * 16 + ln] = (__bf16)g1;
        }
        __syncthreads();

        // ---- phase 2: wave owns H columns [64*wave, 64*wave+64)
#pragma unroll
        for (int kk = 0; kk < KC; kk += 32) {
            v16bf a0 = lds_a16(&as[ln][kk + hs8]);
            v16bf a1 = lds_a16(&as[16 + ln][kk + hs8]);
#pragma unroll
            for (int tt = 0; tt < 4; ++tt) {
                v16bf b = mk_b(wpr_e + (size_t)(wave * 64 + tt * 16 + ln) * I_DIM
                                     + cb + kk + hs8);
                acc[tt][0] = __builtin_amdgcn_wmma_f32_16x16x32_bf16(false, a0, false, b,
                                                                     (short)0, acc[tt][0], false, false);
                acc[tt][1] = __builtin_amdgcn_wmma_f32_16x16x32_bf16(false, a1, false, b,
                                                                     (short)0, acc[tt][1], false, false);
            }
        }
    }

    // Gate-scale and scatter-accumulate (2 fp32 adds per element -> commutative,
    // bitwise deterministic; padded rows contribute exact +0.0f).
#pragma unroll
    for (int tt = 0; tt < 4; ++tt) {
        int col = wave * 64 + tt * 16 + ln;
#pragma unroll
        for (int mg = 0; mg < 2; ++mg) {
#pragma unroll
            for (int r = 0; r < 8; ++r) {
                int row = mg * 16 + r + hs8;
                float v = acc[tt][mg][r] * gate_s[row];
                unsafeAtomicAdd(&out[(size_t)tok_s[row] * H_DIM + col], v);
            }
        }
    }
}

// ---------------------------------------------------------------------------
extern "C" void kernel_launch(void* const* d_in, const int* in_sizes, int n_in,
                              void* d_out, int out_size, void* d_ws, size_t ws_size,
                              hipStream_t stream) {
    const float* x   = (const float*)d_in[0];   // [T, H]
    const float* wg  = (const float*)d_in[1];   // [E, H]
    const float* wfc = (const float*)d_in[2];   // [E, I, H]
    const float* wpr = (const float*)d_in[3];   // [E, H, I]

    float* out    = (float*)d_out;                    // [T, H]
    float* logits = out + (size_t)T_TOK * H_DIM;      // [T, E]

    const size_t W_ELEMS = (size_t)N_EXP * I_DIM * H_DIM;  // per weight tensor
    const size_t LIST_BYTES = (size_t)(2 * N_EXP * T_TOK) * 4 + 256;
    const size_t PRECONV_BYTES = 2 * W_ELEMS * sizeof(__bf16) + LIST_BYTES;

    dim3 ffn_grid(T_TOK / TM, N_EXP);   // worst case: one expert takes all tokens

    if (ws_size >= PRECONV_BYTES) {
        // Preconverted-bf16 path: weights (134 MB) fit the 192 MB L2; FFN inner
        // loop is pure b128 loads + v_wmma (no per-tile converts, no WAR nops).
        __bf16* wfc_b = (__bf16*)d_ws;
        __bf16* wpr_b = wfc_b + W_ELEMS;
        int*    counts    = (int*)(wpr_b + W_ELEMS);
        int*    tok_list  = counts + 16;
        float*  gate_list = (float*)(tok_list + N_EXP * T_TOK);

        moe_zero_kernel<<<(T_TOK * H_DIM / 4) / 256, 256, 0, stream>>>(out, counts);
        moe_cvt_kernel<<<(unsigned)(W_ELEMS / 8 / 256), 256, 0, stream>>>(wfc, wfc_b);
        moe_cvt_kernel<<<(unsigned)(W_ELEMS / 8 / 256), 256, 0, stream>>>(wpr, wpr_b);
        moe_router_kernel<<<T_TOK / WAVES, 256, 0, stream>>>(x, wg, logits, counts,
                                                             tok_list, gate_list);
        moe_ffn_kernel<__bf16><<<ffn_grid, NTHR, 0, stream>>>(x, wfc_b, wpr_b, counts,
                                                              tok_list, gate_list, out);
    } else {
        // Fallback: convert weights in-loop from fp32.
        int*   counts    = (int*)d_ws;
        int*   tok_list  = counts + 16;
        float* gate_list = (float*)(tok_list + N_EXP * T_TOK);

        moe_zero_kernel<<<(T_TOK * H_DIM / 4) / 256, 256, 0, stream>>>(out, counts);
        moe_router_kernel<<<T_TOK / WAVES, 256, 0, stream>>>(x, wg, logits, counts,
                                                             tok_list, gate_list);
        moe_ffn_kernel<float><<<ffn_grid, NTHR, 0, stream>>>(x, wfc, wpr, counts,
                                                             tok_list, gate_list, out);
    }
}
